// NERLSTM_CRF_18743237280656
// MI455X (gfx1250) — compile-verified, matrix operand order checked by reading, not run on hardware
//
#include <hip/hip_runtime.h>
#include <hip/hip_bf16.h>

typedef __attribute__((ext_vector_type(16))) _Float16 v16h;
typedef __attribute__((ext_vector_type(8)))  _Float16 v8h;
typedef __attribute__((ext_vector_type(8)))  float    v8f;
typedef __attribute__((__vector_size__(16))) int      v4i_;

#define B_    256
#define T_    256
#define EMBP  128      // EMB=120 padded to 128 (K multiple of 32)
#define HH_   256      // per-direction hidden
#define G4    1024     // 4*HH (i,f,g,o gate rows)
#define HID_  512
#define K_    13
#define CHD   100
#define SGD   20

// ---- workspace layout (bytes), all offsets 256B aligned; total ~93.3 MB ----
#define OFF_X    ((size_t)0)           // f16 [T][B][128]            16,777,216
#define OFF_WIH  ((size_t)16777216)    // f16 [2][1024][128]            524,288
#define OFF_WHH  ((size_t)17301504)    // f16 [2][1024][256]          1,048,576
#define OFF_HBUF ((size_t)18350080)    // f16 [2dir][2buf][256][256]    524,288
#define OFF_CBUF ((size_t)18874368)    // f32 [2][256][256]             524,288
#define OFF_HALL ((size_t)19398656)    // f32 [T][B][512]            134,217,728
#define OFF_EM   ((size_t)86507520)    // f32 [T][B][13]              3,407,872
#define OFF_HIST ((size_t)89915392)    // i32 [T-1][B][13]            3,394,560

__device__ __forceinline__ float sigmoidf_(float x) {
  return 1.0f / (1.0f + __expf(-x));
}

// ---- CDNA5 async global->LDS copy (ASYNCcnt path), with safe fallback ------
// Builtin prototype (probe-discovered): (global int4*, local int4*, imm, imm)
#if __has_builtin(__builtin_amdgcn_global_load_async_to_lds_b128)
#define ASYNC_CP16(gsrc, ldst)                                               \
  __builtin_amdgcn_global_load_async_to_lds_b128(                            \
      (__attribute__((address_space(1))) v4i_*)(_Float16*)(gsrc),            \
      (__attribute__((address_space(3))) v4i_*)(_Float16*)(ldst), 0, 0)
#else
#define ASYNC_CP16(gsrc, ldst) (*(v8h*)(ldst) = *(const v8h*)(gsrc))
#endif

__device__ __forceinline__ void wait_async0() {
#if __has_builtin(__builtin_amdgcn_s_wait_asynccnt)
  __builtin_amdgcn_s_wait_asynccnt(0);
#else
  asm volatile("s_wait_asynccnt 0" ::: "memory");
#endif
}

// Load a 16-bit WMMA A/B fragment for this lane. p points at (row base + kbase):
// halves [0..7] -> K+0..7, halves [16..23] -> K+16..23 (ISA 16-bit 16x32 layout).
__device__ __forceinline__ v16h load_frag16(const _Float16* __restrict__ p) {
  v8h lo = *(const v8h*)(p);
  v8h hi = *(const v8h*)(p + 16);
  return __builtin_shufflevector(lo, hi, 0,1,2,3,4,5,6,7,8,9,10,11,12,13,14,15);
}

// ---------------------------------------------------------------------------
// Prep: convert/pad weights to f16. idx space covers both w_ih (padded) + w_hh.
__global__ void prep_weights_kernel(const float* __restrict__ w_ih_f,
                                    const float* __restrict__ w_hh_f,
                                    const float* __restrict__ w_ih_b,
                                    const float* __restrict__ w_hh_b,
                                    _Float16* __restrict__ wih,   // [2][1024][128]
                                    _Float16* __restrict__ whh) { // [2][1024][256]
  const int NIH = 2 * G4 * EMBP;          // 262144
  const int NHH = 2 * G4 * HH_;           // 524288
  int idx = blockIdx.x * blockDim.x + threadIdx.x;
  if (idx < NIH) {
    int d   = idx / (G4 * EMBP);
    int rem = idx - d * (G4 * EMBP);
    int row = rem / EMBP;
    int col = rem - row * EMBP;
    const float* src = d ? w_ih_b : w_ih_f;   // [1024][120]
    float v = (col < 120) ? src[row * 120 + col] : 0.0f;
    wih[idx] = (_Float16)v;
  } else if (idx < NIH + NHH) {
    int j   = idx - NIH;
    int d   = j / (G4 * HH_);
    int rem = j - d * (G4 * HH_);
    const float* src = d ? w_hh_b : w_hh_f;   // [1024][256] contiguous
    whh[j] = (_Float16)src[rem];
  }
}

// Embedding gather + concat + pad -> x f16 [T][B][128]
__global__ void embed_kernel(const int* __restrict__ char_ids,   // [B][T]
                             const int* __restrict__ seg_ids,    // [B][T]
                             const float* __restrict__ char_emb, // [V][100]
                             const float* __restrict__ seg_emb,  // [5][20]
                             _Float16* __restrict__ xf16) {
  int idx = blockIdx.x * blockDim.x + threadIdx.x;   // T*B*128
  if (idx >= T_ * B_ * EMBP) return;
  int t   = idx / (B_ * EMBP);
  int rem = idx - t * (B_ * EMBP);
  int b   = rem / EMBP;
  int k   = rem - b * EMBP;
  float v = 0.0f;
  if (k < CHD) {
    int id = char_ids[b * T_ + t];
    v = char_emb[(size_t)id * CHD + k];
  } else if (k < CHD + SGD) {
    int id = seg_ids[b * T_ + t];
    v = seg_emb[id * SGD + (k - CHD)];
  }
  xf16[idx] = (_Float16)v;
}

// Zero h ping-pong buffers and c state (deterministic per call).
__global__ void init_state_kernel(_Float16* __restrict__ hbuf,  // 2*2*65536 halves
                                  float* __restrict__ cbuf) {   // 2*65536 floats
  int idx = blockIdx.x * blockDim.x + threadIdx.x;
  const int NH = 2 * 2 * B_ * HH_;
  const int NC = 2 * B_ * HH_;
  if (idx < NH)            hbuf[idx] = (_Float16)0.0f;
  else if (idx < NH + NC)  cbuf[idx - NH] = 0.0f;
}

// ---------------------------------------------------------------------------
// One LSTM timestep, both directions (blockIdx.y). All 8 waves of a block
// share the same 16-row batch tile (m0), so the A-operands (x slice + h_prev
// slice) are staged once per block into LDS via the CDNA5 async global->LDS
// path (ASYNCcnt), then WMMA A-fragments are fed from LDS (ds_load_b128)
// while each wave streams its own B (weight) fragments from L2-resident
// global memory. Each wave owns a 16(batch) x 16(hidden) tile and accumulates
// all 4 gate tiles with WMMA over K = 128 (x @ W_ih^T) + 256 (h @ W_hh^T),
// then applies the cell update in-register on the accumulator layout.
__global__ void __launch_bounds__(256)
lstm_step_kernel(const _Float16* __restrict__ xf16,  // [T][B][128]
                 const _Float16* __restrict__ wih,   // [2][1024][128]
                 const _Float16* __restrict__ whh,   // [2][1024][256]
                 const float* __restrict__ b_f,      // [1024]
                 const float* __restrict__ b_b,      // [1024]
                 _Float16* __restrict__ hbuf,        // [2][2][256][256]
                 float* __restrict__ cbuf,           // [2][256][256]
                 float* __restrict__ hall,           // [T][B][512]
                 int t) {
  const int d    = blockIdx.y;                       // 0=fwd, 1=bwd
  const int tt   = d ? (T_ - 1 - t) : t;             // input time index
  const int w    = threadIdx.x >> 5;                 // wave in block (0..7)
  const int lane = threadIdx.x & 31;
  const int ln   = lane & 15;
  const int kb   = (lane < 16) ? 0 : 8;              // 16-bit frag K base
  const int m0   = (blockIdx.x >> 1) << 4;           // batch tile (per block)
  const int h0   = (((blockIdx.x & 1) << 3) | w) << 4; // hidden tile (per wave)

  const _Float16* hcur = hbuf + ((size_t)d * 2 + (t & 1)) * (B_ * HH_);
  _Float16*       hnxt = hbuf + ((size_t)d * 2 + ((t + 1) & 1)) * (B_ * HH_);

  // ---- stage shared A tiles (x: 16x128, h_prev: 16x256) into LDS ----------
  __shared__ _Float16 lx[16 * EMBP];   // 4 KB
  __shared__ _Float16 lh[16 * HH_];    // 8 KB
  {
    const int tid = threadIdx.x;
    // x tile: 256 x 16B chunks, one per thread
    {
      const int row = tid >> 4;              // 16 chunks per row
      const int col = (tid & 15) << 3;       // in halves
      const _Float16* g = xf16 + ((size_t)tt * B_ + m0 + row) * EMBP + col;
      ASYNC_CP16(g, lx + row * EMBP + col);
    }
    // h tile: 512 x 16B chunks, two per thread
    #pragma unroll
    for (int j = 0; j < 2; ++j) {
      const int chunk = tid + (j << 8);
      const int row = chunk >> 5;            // 32 chunks per row
      const int col = (chunk & 31) << 3;
      const _Float16* g = hcur + (size_t)(m0 + row) * HH_ + col;
      ASYNC_CP16(g, lh + row * HH_ + col);
    }
    wait_async0();
    __syncthreads();
  }

  const _Float16* lxrow = lx + (size_t)ln * EMBP + kb;
  const _Float16* lhrow = lh + (size_t)ln * HH_ + kb;
  const _Float16* wihd  = wih + (size_t)d * (G4 * EMBP);
  const _Float16* whhd  = whh + (size_t)d * (G4 * HH_);

  v8f acc[4];
  #pragma unroll
  for (int g = 0; g < 4; ++g) {
    v8f z = {0.f, 0.f, 0.f, 0.f, 0.f, 0.f, 0.f, 0.f};
    acc[g] = z;
  }

  // x contribution: K = 128 (A from LDS, B from global)
  #pragma unroll
  for (int k0 = 0; k0 < EMBP; k0 += 32) {
    v16h a = load_frag16(lxrow + k0);
    #pragma unroll
    for (int g = 0; g < 4; ++g) {
      const _Float16* wr = wihd + (size_t)(g * HH_ + h0 + ln) * EMBP + k0 + kb;
      v16h bfr = load_frag16(wr);
      acc[g] = __builtin_amdgcn_wmma_f32_16x16x32_f16(
          false, a, false, bfr, (short)0, acc[g], false, false);
    }
  }
  // recurrent contribution: K = 256 (A from LDS, B from global)
  #pragma unroll
  for (int k0 = 0; k0 < HH_; k0 += 32) {
    v16h a = load_frag16(lhrow + k0);
    #pragma unroll
    for (int g = 0; g < 4; ++g) {
      const _Float16* wr = whhd + (size_t)(g * HH_ + h0 + ln) * HH_ + k0 + kb;
      v16h bfr = load_frag16(wr);
      acc[g] = __builtin_amdgcn_wmma_f32_16x16x32_f16(
          false, a, false, bfr, (short)0, acc[g], false, false);
    }
  }

  const float* bias = d ? b_b : b_f;
  const int ncol = h0 + ln;                 // hidden column this lane owns
  const float bi = bias[0 * HH_ + ncol];
  const float bf = bias[1 * HH_ + ncol];
  const float bg = bias[2 * HH_ + ncol];
  const float bo = bias[3 * HH_ + ncol];
  float* crow = cbuf + (size_t)d * (B_ * HH_);

  // accumulator layout: acc[g][i] = element (m = i + 8*(lane>=16), n = lane&15)
  #pragma unroll
  for (int i = 0; i < 8; ++i) {
    const int m  = i + ((lane >> 4) << 3);
    const int bb = m0 + m;
    const float gi = acc[0][i] + bi;
    const float gf = acc[1][i] + bf;
    const float gg = acc[2][i] + bg;
    const float go = acc[3][i] + bo;
    const size_t ci = (size_t)bb * HH_ + ncol;
    const float cp = crow[ci];
    const float cn = sigmoidf_(gf) * cp + sigmoidf_(gi) * tanhf(gg);
    const float hn = sigmoidf_(go) * tanhf(cn);
    crow[ci] = cn;
    hall[((size_t)tt * B_ + bb) * HID_ + d * HH_ + ncol] = hn;
    hnxt[ci] = (_Float16)hn;
  }
}

// ---------------------------------------------------------------------------
// Emissions: em[t,b,k] = h_all[t,b,:] . w_out[k,:] + b_out[k]  (K=13, tiny N)
__global__ void emission_kernel(const float* __restrict__ hall,  // [T][B][512]
                                const float* __restrict__ w_out, // [13][512]
                                const float* __restrict__ b_out, // [13]
                                float* __restrict__ em) {        // [T][B][13]
  int idx = blockIdx.x * blockDim.x + threadIdx.x;  // T*B*13
  if (idx >= T_ * B_ * K_) return;
  int t   = idx / (B_ * K_);
  int rem = idx - t * (B_ * K_);
  int b   = rem / K_;
  int k   = rem - b * K_;
  const float* h = hall + ((size_t)t * B_ + b) * HID_;
  const float* w = w_out + k * HID_;
  float s = b_out[k];
  #pragma unroll 8
  for (int j = 0; j < HID_; ++j) s += h[j] * w[j];
  em[idx] = s;
}

// Viterbi decode: one block per batch row, 13 active lanes, sequential over T.
__global__ void viterbi_kernel(const float* __restrict__ em,    // [T][B][13]
                               const float* __restrict__ trans, // [13][13]
                               const float* __restrict__ st,    // [13]
                               const float* __restrict__ en,    // [13]
                               int* __restrict__ hist,          // [T-1][B][13]
                               float* __restrict__ out) {       // path B*T + score B
  const int b = blockIdx.x;
  const int k = threadIdx.x;
  __shared__ float sc[K_];
  __shared__ float tr[K_ * K_];
  for (int j = k; j < K_ * K_; j += 32) tr[j] = trans[j];
  if (k < K_) sc[k] = st[k] + em[(size_t)b * K_ + k];
  __syncthreads();
  for (int t = 1; t < T_; ++t) {
    float best = -3.0e38f;
    int arg = 0;
    if (k < K_) {
      #pragma unroll
      for (int kp = 0; kp < K_; ++kp) {
        float v = sc[kp] + tr[kp * K_ + k];
        if (v > best) { best = v; arg = kp; }
      }
      hist[((size_t)(t - 1) * B_ + b) * K_ + k] = arg;
      best += em[((size_t)t * B_ + b) * K_ + k];
    }
    __syncthreads();
    if (k < K_) sc[k] = best;
    __syncthreads();
  }
  if (k == 0) {
    float bs = -3.0e38f;
    int bl = 0;
    for (int kk = 0; kk < K_; ++kk) {
      float v = sc[kk] + en[kk];
      if (v > bs) { bs = v; bl = kk; }
    }
    out[(size_t)B_ * T_ + b] = bs;          // best_score[b]
    int tag = bl;
    out[(size_t)b * T_ + (T_ - 1)] = (float)tag;
    for (int t = T_ - 2; t >= 0; --t) {
      tag = hist[((size_t)t * B_ + b) * K_ + tag];
      out[(size_t)b * T_ + t] = (float)tag;
    }
  }
}

// ---------------------------------------------------------------------------
extern "C" void kernel_launch(void* const* d_in, const int* in_sizes, int n_in,
                              void* d_out, int out_size, void* d_ws, size_t ws_size,
                              hipStream_t stream) {
  const int*   char_ids = (const int*)d_in[0];
  const int*   seg_ids  = (const int*)d_in[1];
  const float* char_emb = (const float*)d_in[2];
  const float* seg_emb  = (const float*)d_in[3];
  const float* w_ih_f   = (const float*)d_in[4];
  const float* w_hh_f   = (const float*)d_in[5];
  const float* b_f      = (const float*)d_in[6];
  const float* w_ih_b   = (const float*)d_in[7];
  const float* w_hh_b   = (const float*)d_in[8];
  const float* b_b      = (const float*)d_in[9];
  const float* w_out    = (const float*)d_in[10];
  const float* b_out    = (const float*)d_in[11];
  const float* trans    = (const float*)d_in[12];
  const float* start_t  = (const float*)d_in[13];
  const float* end_t    = (const float*)d_in[14];

  char* ws = (char*)d_ws;
  _Float16* xf16 = (_Float16*)(ws + OFF_X);
  _Float16* wih  = (_Float16*)(ws + OFF_WIH);
  _Float16* whh  = (_Float16*)(ws + OFF_WHH);
  _Float16* hbuf = (_Float16*)(ws + OFF_HBUF);
  float*    cbuf = (float*)(ws + OFF_CBUF);
  float*    hall = (float*)(ws + OFF_HALL);
  float*    em   = (float*)(ws + OFF_EM);
  int*      hist = (int*)(ws + OFF_HIST);
  float*    out  = (float*)d_out;

  // 1) weight convert/pad (f32 -> f16)
  {
    int n = 2 * G4 * EMBP + 2 * G4 * HH_;
    prep_weights_kernel<<<(n + 255) / 256, 256, 0, stream>>>(
        w_ih_f, w_hh_f, w_ih_b, w_hh_b, wih, whh);
  }
  // 2) embedding gather -> x f16
  {
    int n = T_ * B_ * EMBP;
    embed_kernel<<<(n + 255) / 256, 256, 0, stream>>>(
        char_ids, seg_ids, char_emb, seg_emb, xf16);
  }
  // 3) zero h/c state
  {
    int n = 2 * 2 * B_ * HH_ + 2 * B_ * HH_;
    init_state_kernel<<<(n + 255) / 256, 256, 0, stream>>>(hbuf, cbuf);
  }
  // 4) 256 recurrent steps, fwd+bwd concurrently via blockIdx.y
  for (int t = 0; t < T_; ++t) {
    lstm_step_kernel<<<dim3(32, 2), 256, 0, stream>>>(
        xf16, wih, whh, b_f, b_b, hbuf, cbuf, hall, t);
  }
  // 5) emissions
  {
    int n = T_ * B_ * K_;
    emission_kernel<<<(n + 255) / 256, 256, 0, stream>>>(hall, w_out, b_out, em);
  }
  // 6) Viterbi decode + backtrace -> d_out (path B*T as float, then score B)
  viterbi_kernel<<<B_, 32, 0, stream>>>(em, trans, start_t, end_t, hist, out);
}
